// DGCNN_62491774157184
// MI455X (gfx1250) — matrix operand order repaired and checked by reading, not compiled, not run on hardware
//
#include <hip/hip_runtime.h>
#include <hip/hip_bf16.h>

// ---------------------------------------------------------------------------
// DGCNN / SEAL forward for MI455X (gfx1250, wave32, WMMA).
//   - GCN GEMMs and the MLP GEMM use v_wmma_f32_16x16x32_f16 (f32 accum),
//     templated on KSTEPS so B-fragment addresses are immediate offsets.
//   - Edge aggregation via float atomics (accumulator is L2-resident),
//     per-wave edge indices broadcast from lane 0.
//   - Sort-pool via per-graph O(n^2) stable rank selection.
// ---------------------------------------------------------------------------

typedef __attribute__((ext_vector_type(16))) _Float16 v16h;
typedef __attribute__((ext_vector_type(8)))  float    v8f;

struct alignas(16) U128 { unsigned x, y, z, w; };
struct alignas(16) F4   { float x, y, z, w; };

// ---------------- degree / normalization -----------------------------------

__global__ void init_deg_k(float* deg, int N) {
    int i = blockIdx.x * blockDim.x + threadIdx.x;
    if (i < N) deg[i] = 1.0f;                       // self-loop
}

__global__ void deg_edges_k(const long long* __restrict__ ei, float* deg, int E) {
    int e = blockIdx.x * blockDim.x + threadIdx.x;
    if (e < E) atomicAdd(deg + (int)ei[(size_t)E + e], 1.0f);
}

__global__ void rsqrt_k(float* d, int N) {
    int i = blockIdx.x * blockDim.x + threadIdx.x;
    if (i < N) d[i] = rsqrtf(d[i]);                 // deg >= 1 always
}

__global__ void f32_to_f16_k(const float* __restrict__ in, _Float16* __restrict__ out, int n) {
    int i = blockIdx.x * blockDim.x + threadIdx.x;
    if (i < n) out[i] = (_Float16)in[i];
}

// ---------------- WMMA GEMM (C = A[f16, MxK] * Wpacked -> f32 Mx128) --------
// B fragments pre-packed in contiguous-k lane order (k = 16*(lane/16) + elem): [ntile(8)][kstep][lane(32)][8 u32]

__global__ void pack_b_wmma(const float* __restrict__ W, unsigned* __restrict__ out,
                            int Nw, int ksteps) {
    int idx = blockIdx.x * blockDim.x + threadIdx.x;
    int total = 8 * ksteps * 256;
    if (idx >= total) return;
    int j    = idx & 7;
    int lane = (idx >> 3) & 31;
    int ks   = (idx >> 8) % ksteps;
    int nt   = idx / (ksteps * 256);
    int n    = nt * 16 + (lane & 15);
    int kb   = ks * 32 + ((lane & 16) ? 16 : 0);
    union { _Float16 h[2]; unsigned u; } p;
    p.h[0] = (_Float16)W[(size_t)(kb + 2 * j)     * Nw + n];
    p.h[1] = (_Float16)W[(size_t)(kb + 2 * j + 1) * Nw + n];
    out[idx] = p.u;
}

template <int KSTEPS>
__global__ __launch_bounds__(256) void gemm_wmma_f16(
    const _Float16* __restrict__ A, const unsigned* __restrict__ Bfrag,
    float* __restrict__ C, int M) {
    const int lane = threadIdx.x & 31;
    const int wave = threadIdx.x >> 5;
    const int m0   = (blockIdx.x * 8 + wave) * 16;
    if (m0 >= M) return;                            // uniform per wave
    const int lr = lane & 15;
    const int hi = lane >> 4;
    const _Float16* arow = A + (size_t)(m0 + lr) * (KSTEPS * 32) + (hi ? 16 : 0);
    // per-lane B base in U128 units; nt/ks offsets are compile-time constants
    const U128* bbase = (const U128*)Bfrag + lane * 2;

    v8f acc[8];
#pragma unroll
    for (int nt = 0; nt < 8; ++nt) acc[nt] = (v8f){0, 0, 0, 0, 0, 0, 0, 0};

    for (int ks = 0; ks < KSTEPS; ++ks) {
        union { v16h v; U128 q[2]; } a;
        const _Float16* ab = arow + ks * 32;
        a.q[0] = *(const U128*)(ab);
        a.q[1] = *(const U128*)(ab + 8);
        const U128* bp = bbase + ks * 64;
        union { v16h v; U128 q[2]; } b[8];
#pragma unroll
        for (int nt = 0; nt < 8; ++nt) {
            b[nt].q[0] = bp[nt * (KSTEPS * 64)];
            b[nt].q[1] = bp[nt * (KSTEPS * 64) + 1];
        }
#pragma unroll
        for (int nt = 0; nt < 8; ++nt)
            acc[nt] = __builtin_amdgcn_wmma_f32_16x16x32_f16(
                false, a.v, false, b[nt].v, (short)0, acc[nt], false, false);
    }
#pragma unroll
    for (int nt = 0; nt < 8; ++nt) {
        const int col = nt * 16 + lr;
#pragma unroll
        for (int r = 0; r < 8; ++r) {
            int row = m0 + r + (hi ? 8 : 0);
            C[(size_t)row * 128 + col] = acc[nt][r];
        }
    }
}

// ---------------- GCN aggregation -------------------------------------------

__global__ void selfloop_init_k(const float* __restrict__ tmp, const float* __restrict__ dinv,
                                float* __restrict__ agg, int total) {
    int i = blockIdx.x * blockDim.x + threadIdx.x;
    if (i >= total) return;
    int n = i >> 7;
    float dv = dinv[n];
    agg[i] = tmp[i] * dv * dv;
}

__global__ __launch_bounds__(256) void gcn_scatter_k(
    const long long* __restrict__ ei, const float* __restrict__ dinv,
    const float* __restrict__ tmp, float* __restrict__ agg, int E) {
    long long gid = (long long)blockIdx.x * 256 + threadIdx.x;
    int e = (int)(gid >> 5);
    if (e >= E) return;                             // uniform per wave
    int lane = (int)gid & 31;
    int s = 0, d = 0;
    float norm = 0.0f;
    if (lane == 0) {
        s = (int)ei[e];
        d = (int)ei[(size_t)E + e];
        norm = dinv[s] * dinv[d];
    }
    s    = __shfl(s, 0);
    d    = __shfl(d, 0);
    norm = __shfl(norm, 0);
    int c0 = lane * 4;
    F4 v = *(const F4*)(tmp + (size_t)s * 128 + c0);
    float* dp = agg + (size_t)d * 128 + c0;
    atomicAdd(dp + 0, v.x * norm);
    atomicAdd(dp + 1, v.y * norm);
    atomicAdd(dp + 2, v.z * norm);
    atomicAdd(dp + 3, v.w * norm);
}

__global__ void finalize_layer_k(const float* __restrict__ agg, const float* __restrict__ bias,
                                 float* __restrict__ feat, _Float16* __restrict__ hbf,
                                 int total, int coloff) {
    int i = blockIdx.x * blockDim.x + threadIdx.x;
    if (i >= total) return;
    int n = i >> 7, c = i & 127;
    float h = tanhf(agg[i] + bias[c]);
    feat[(size_t)n * 385 + coloff + c] = h;
    hbf[i] = (_Float16)h;
}

// ---------------- layer 3 (H -> 1) ------------------------------------------

__global__ void dot_w3_k(const _Float16* __restrict__ hbf, const float* __restrict__ Wg3,
                         float* __restrict__ tmp1, int N) {
    int gid = blockIdx.x * blockDim.x + threadIdx.x;
    int node = gid >> 5, lane = gid & 31;
    if (node >= N) return;
    float s = 0.0f;
#pragma unroll
    for (int t = 0; t < 4; ++t) {
        int c = lane + 32 * t;
        s += (float)hbf[(size_t)node * 128 + c] * Wg3[c];
    }
    for (int off = 16; off > 0; off >>= 1) s += __shfl_down(s, off);
    if (lane == 0) tmp1[node] = s;
}

__global__ void selfloop1_k(const float* tmp1, const float* dinv, float* agg1, int N) {
    int n = blockIdx.x * blockDim.x + threadIdx.x;
    if (n < N) { float dv = dinv[n]; agg1[n] = tmp1[n] * dv * dv; }
}

__global__ void scatter1_k(const long long* __restrict__ ei, const float* __restrict__ dinv,
                           const float* __restrict__ tmp1, float* __restrict__ agg1, int E) {
    int e = blockIdx.x * blockDim.x + threadIdx.x;
    if (e >= E) return;
    int s = (int)ei[e];
    int d = (int)ei[(size_t)E + e];
    atomicAdd(agg1 + d, tmp1[s] * dinv[s] * dinv[d]);
}

__global__ void finalize_scalar_k(const float* __restrict__ agg1, const float* __restrict__ bg3,
                                  float* __restrict__ feat, int N) {
    int n = blockIdx.x * blockDim.x + threadIdx.x;
    if (n < N) feat[(size_t)n * 385 + 384] = tanhf(agg1[n] + bg3[0]);
}

// ---------------- batch bookkeeping + sort-pool -----------------------------

__global__ void zero_i32_k(int* p, int n) {
    int i = blockIdx.x * blockDim.x + threadIdx.x;
    if (i < n) p[i] = 0;
}

__global__ void count_batch_k(const long long* __restrict__ batch, int* counts, int N) {
    int n = blockIdx.x * blockDim.x + threadIdx.x;
    if (n < N) atomicAdd(counts + (int)batch[n], 1);
}

__global__ void scan_counts_k(const int* counts, int* starts, int B) {
    if (blockIdx.x == 0 && threadIdx.x == 0) {
        int acc = 0;
        for (int b = 0; b < B; ++b) { starts[b] = acc; acc += counts[b]; }
    }
}

__global__ __launch_bounds__(256) void sortpool_k(
    const float* __restrict__ feat, const int* __restrict__ counts,
    const int* __restrict__ starts, float* __restrict__ pooled) {
    __shared__ float vals[1024];
    __shared__ int sel[100];
    int b = blockIdx.x;
    int start = starts[b];
    int cnt = counts[b];
    if (cnt > 1024) cnt = 1024;
    for (int i = threadIdx.x; i < cnt; i += 256)
        vals[i] = feat[(size_t)(start + i) * 385 + 384];
    __syncthreads();
    for (int i = threadIdx.x; i < cnt; i += 256) {
        float v = vals[i];
        int r = 0;
        for (int j = 0; j < cnt; ++j) {
            float u = vals[j];
            r += (u > v) || (u == v && j < i);      // stable, descending
        }
        if (r < 100) sel[r] = start + i;
    }
    __syncthreads();
    for (int idx = threadIdx.x; idx < 100 * 385; idx += 256) {
        int r = idx / 385, c = idx % 385;
        float o = 0.0f;
        if (r < cnt) o = feat[(size_t)sel[r] * 385 + c];
        pooled[(size_t)(b * 100 + r) * 385 + c] = o;
    }
}

// ---------------- fused conv1 / maxpool / conv2 head ------------------------

__global__ __launch_bounds__(256) void cnn_head_k(
    const float* __restrict__ pooled, const float* __restrict__ W1, const float* __restrict__ b1,
    const float* __restrict__ W2, const float* __restrict__ b2,
    _Float16* __restrict__ flat16) {
    __shared__ float w1s[16 * 385];
    __shared__ float h1[16 * 100];
    __shared__ float h1p[16 * 50];
    __shared__ float w2s[32 * 16 * 5];
    int b = blockIdx.x;
    for (int i = threadIdx.x; i < 16 * 385; i += 256) w1s[i] = W1[i];
    for (int i = threadIdx.x; i < 32 * 16 * 5; i += 256) w2s[i] = W2[i];
    __syncthreads();
    const float* pb = pooled + (size_t)b * 100 * 385;
    for (int idx = threadIdx.x; idx < 1600; idx += 256) {
        int c = idx / 100, j = idx % 100;
        const float* row = pb + (size_t)j * 385;
        const float* w = w1s + c * 385;
        float s = b1[c];
        for (int d = 0; d < 385; ++d) s += w[d] * row[d];
        h1[c * 100 + j] = fmaxf(s, 0.0f);
    }
    __syncthreads();
    for (int idx = threadIdx.x; idx < 800; idx += 256) {
        int c = idx / 50, j = idx % 50;
        h1p[c * 50 + j] = fmaxf(h1[c * 100 + 2 * j], h1[c * 100 + 2 * j + 1]);
    }
    __syncthreads();
    for (int idx = threadIdx.x; idx < 32 * 46; idx += 256) {
        int oc = idx / 46, t = idx % 46;
        float s = b2[oc];
#pragma unroll
        for (int ic = 0; ic < 16; ++ic)
#pragma unroll
            for (int kw = 0; kw < 5; ++kw)
                s += h1p[ic * 50 + t + kw] * w2s[oc * 80 + ic * 5 + kw];
        flat16[(size_t)b * 1472 + idx] = (_Float16)fmaxf(s, 0.0f);
    }
}

// ---------------- final MLP reduction ---------------------------------------

__global__ __launch_bounds__(128) void mlp_final_k(
    const float* __restrict__ gout, const float* __restrict__ bm1,
    const float* __restrict__ Wm2, const float* __restrict__ bm2,
    float* __restrict__ out) {
    __shared__ float red[128];
    int b = blockIdx.x, j = threadIdx.x;
    red[j] = fmaxf(gout[(size_t)b * 128 + j] + bm1[j], 0.0f) * Wm2[j];
    __syncthreads();
    for (int s = 64; s > 0; s >>= 1) {
        if (j < s) red[j] += red[j + s];
        __syncthreads();
    }
    if (j == 0) out[b] = red[0] + bm2[0];
}

// ---------------------------------------------------------------------------

extern "C" void kernel_launch(void* const* d_in, const int* in_sizes, int n_in,
                              void* d_out, int out_size, void* d_ws, size_t ws_size,
                              hipStream_t stream) {
    const int N = in_sizes[0] / 128;
    const int E = in_sizes[1] / 2;
    const int B = out_size;                          // 512

    const float*     x     = (const float*)d_in[0];
    const long long* ei    = (const long long*)d_in[1];
    const long long* batch = (const long long*)d_in[2];
    const float* Wg[3] = { (const float*)d_in[4], (const float*)d_in[6], (const float*)d_in[8] };
    const float* bg[3] = { (const float*)d_in[5], (const float*)d_in[7], (const float*)d_in[9] };
    const float* Wg3 = (const float*)d_in[10];
    const float* bg3 = (const float*)d_in[11];
    const float* W1  = (const float*)d_in[12];
    const float* b1  = (const float*)d_in[13];
    const float* W2  = (const float*)d_in[14];
    const float* b2  = (const float*)d_in[15];
    const float* Wm1 = (const float*)d_in[16];
    const float* bm1 = (const float*)d_in[17];
    const float* Wm2 = (const float*)d_in[18];
    const float* bm2 = (const float*)d_in[19];
    float* out = (float*)d_out;

    // ---- workspace layout (256B aligned slices) ----
    char* cur = (char*)d_ws;
    auto alloc = [&](size_t bytes) -> void* {
        void* p = cur;
        cur += (bytes + 255) & ~(size_t)255;
        return p;
    };
    float*     dinv   = (float*)alloc((size_t)N * 4);
    _Float16*  hbf    = (_Float16*)alloc((size_t)N * 128 * 2);
    float*     tmp    = (float*)alloc((size_t)N * 128 * 4);
    float*     agg    = (float*)alloc((size_t)N * 128 * 4);
    float*     feat   = (float*)alloc((size_t)N * 385 * 4);
    float*     tmp1   = (float*)alloc((size_t)N * 4);
    float*     agg1   = (float*)alloc((size_t)N * 4);
    int*       counts = (int*)alloc((size_t)B * 4);
    int*       starts = (int*)alloc((size_t)B * 4);
    unsigned*  bfragG = (unsigned*)alloc(3 * 8 * 4 * 32 * 8 * 4);   // 3 x 32KB
    unsigned*  bfragM = (unsigned*)alloc((size_t)8 * 46 * 32 * 8 * 4);
    _Float16*  flat16 = (_Float16*)alloc((size_t)B * 1472 * 2);
    float*     gout   = (float*)alloc((size_t)B * 128 * 4);
    float*     pooled = tmp;   // alias: tmp+agg (contiguous, 102MB) free by pool time

    const int T = 256;
    const int totNF = N * 128;

    // ---- degree + symmetric normalization ----
    init_deg_k<<<(N + T - 1) / T, T, 0, stream>>>(dinv, N);
    deg_edges_k<<<(E + T - 1) / T, T, 0, stream>>>(ei, dinv, E);
    rsqrt_k<<<(N + T - 1) / T, T, 0, stream>>>(dinv, N);

    // ---- pack weights into WMMA B-fragment layout ----
    for (int l = 0; l < 3; ++l)
        pack_b_wmma<<<(8 * 4 * 256 + T - 1) / T, T, 0, stream>>>(Wg[l], bfragG + l * 8192, 128, 4);
    pack_b_wmma<<<(8 * 46 * 256 + T - 1) / T, T, 0, stream>>>(Wm1, bfragM, 128, 46);

    // ---- input -> f16 ----
    f32_to_f16_k<<<(totNF + T - 1) / T, T, 0, stream>>>(x, hbf, totNF);

    // ---- GCN layers 0..2 ----
    const int mtiles = N / 16;                       // N = 100000 -> 6250
    const int ggrid  = (mtiles + 7) / 8;
    for (int l = 0; l < 3; ++l) {
        gemm_wmma_f16<4><<<ggrid, 256, 0, stream>>>(hbf, bfragG + l * 8192, tmp, N);
        selfloop_init_k<<<(totNF + T - 1) / T, T, 0, stream>>>(tmp, dinv, agg, totNF);
        gcn_scatter_k<<<(int)(((long long)E * 32 + 255) / 256), 256, 0, stream>>>(ei, dinv, tmp, agg, E);
        finalize_layer_k<<<(totNF + T - 1) / T, T, 0, stream>>>(agg, bg[l], feat, hbf, totNF, l * 128);
    }

    // ---- GCN layer 3 (H -> 1) ----
    dot_w3_k<<<((N * 32) + T - 1) / T, T, 0, stream>>>(hbf, Wg3, tmp1, N);
    selfloop1_k<<<(N + T - 1) / T, T, 0, stream>>>(tmp1, dinv, agg1, N);
    scatter1_k<<<(E + T - 1) / T, T, 0, stream>>>(ei, dinv, tmp1, agg1, E);
    finalize_scalar_k<<<(N + T - 1) / T, T, 0, stream>>>(agg1, bg3, feat, N);

    // ---- sort pooling ----
    zero_i32_k<<<(B + T - 1) / T, T, 0, stream>>>(counts, B);
    count_batch_k<<<(N + T - 1) / T, T, 0, stream>>>(batch, counts, N);
    scan_counts_k<<<1, 1, 0, stream>>>(counts, starts, B);
    sortpool_k<<<B, 256, 0, stream>>>(feat, counts, starts, pooled);

    // ---- CNN head -> flat (f16) ----
    cnn_head_k<<<B, 256, 0, stream>>>(pooled, W1, b1, W2, b2, flat16);

    // ---- MLP hidden GEMM via WMMA (512 x 1472 x 128) ----
    gemm_wmma_f16<46><<<(B / 16 + 7) / 8, 256, 0, stream>>>(flat16, bfragM, gout, B);

    // ---- final reduction ----
    mlp_final_k<<<B, 128, 0, stream>>>(gout, bm1, Wm2, bm2, out);

    (void)n_in; (void)ws_size;
}
